// Decoder_34273839022148
// MI455X (gfx1250) — compile-verified, matrix operand order checked by reading, not run on hardware
//
#include <hip/hip_runtime.h>

#define VOCABN 34
#define HIDN 512
#define EMBN 128
#define KSZN 128
#define VSZN 128
#define TLEN 1024
#define NBATCH 128
#define LSTEPS 200
#define NTHREADS 512

typedef unsigned short u16;
typedef __attribute__((ext_vector_type(16))) __bf16 v16bf;
typedef __attribute__((ext_vector_type(8)))  float  v8f;

__device__ __forceinline__ u16 f2bf(float f) {
  union { float f; unsigned u; } x; x.f = f;
  return (u16)((x.u + 0x7FFFu + ((x.u >> 16) & 1u)) >> 16);
}
__device__ __forceinline__ float bf2f(u16 b) {
  union { unsigned u; float f; } x; x.u = ((unsigned)b) << 16;
  return x.f;
}
__device__ __forceinline__ float sigm(float x) { return 1.f / (1.f + __expf(-x)); }
__device__ __forceinline__ float tanh_fast(float x) {
  float ax = fabsf(x);
  float e = __expf(2.f * ax);
  float r = 1.f - 2.f / (e + 1.f);
  return copysignf(r, x);
}

// One 16x16 output tile: D += A(16xK) * B(Kx16) over KT k-steps of 32 (bf16 WMMA).
__device__ __forceinline__ v8f wmma_tile(const u16* Abuf, const u16* Wsw,
                                         int nt, int KT, int lane) {
  v8f acc = {};
  const u16* wb = Wsw + (((size_t)nt * KT) << 9) + lane * 16;
  const u16* ab = Abuf + lane * 16;
  #pragma unroll 2
  for (int kt = 0; kt < KT; ++kt) {
    v16bf a = *reinterpret_cast<const v16bf*>(ab + (kt << 9));
    v16bf b = *reinterpret_cast<const v16bf*>(wb + (kt << 9));
    acc = __builtin_amdgcn_wmma_f32_16x16x32_bf16(false, a, false, b, (short)0, acc,
                                                  false, false);
  }
  return acc;
}

// Pack f32 activations [m][k] (up to 3 concatenated LDS sources) into WMMA A-operand
// lane layout: element (m,k): m = lane&15, k = 32*kt + 16*(j/8) + 8*(lane/16) + (j%8).
__device__ __forceinline__ void swizzleA(u16* Abuf,
    const float* s0, int l0, const float* s1, int l1, const float* s2, int l2,
    int Ktot, int tid) {
  int total = (Ktot >> 5) << 9;  // (Ktot/32) * 32 lanes * 16
  for (int id = tid; id < total; id += NTHREADS) {
    int j = id & 15, lane = (id >> 4) & 31, kt = id >> 9;
    int m = lane & 15;
    int k = (kt << 5) + ((j >> 3) << 4) + ((lane >> 4) << 3) + (j & 7);
    float v;
    if (k < l0)            v = s0[m * l0 + k];
    else if (k < l0 + l1)  v = s1[m * l1 + (k - l0)];
    else                   v = s2[m * l2 + (k - l0 - l1)];
    Abuf[id] = f2bf(v);
  }
}

// Masked softmax attention; one wave owns one batch row over the full T range.
// Pure intra-wave reductions -> no barriers, no cross-wave LDS state.
__device__ __forceinline__ void attend(
    const u16* __restrict__ Kbf, const u16* __restrict__ Vbf,
    const int* __restrict__ redlen, const float* qbuf,
    float* energy, float* ctx, int nb, int wave, int lane) {
  int b = wave;                 // 16 waves, 16 batch rows
  int n = nb + b;
  int red = redlen[n];

  float mx = -3.0e38f;
  for (int t0 = 0; t0 < TLEN; t0 += 32) {
    int t = t0 + lane;
    float e = -1.0e9f;
    if (t < red) {
      e = 0.f;
      const u16* kp = Kbf + ((size_t)t * NBATCH + n) * KSZN;
      __builtin_prefetch(kp + 32 * NBATCH * KSZN, 0, 1);
      #pragma unroll
      for (int kk = 0; kk < KSZN; kk += 8) {
        uint4 pk = *reinterpret_cast<const uint4*>(kp + kk);
        float4 q0 = *reinterpret_cast<const float4*>(&qbuf[b * KSZN + kk]);
        float4 q1 = *reinterpret_cast<const float4*>(&qbuf[b * KSZN + kk + 4]);
        e += bf2f((u16)(pk.x & 0xffff)) * q0.x; e += bf2f((u16)(pk.x >> 16)) * q0.y;
        e += bf2f((u16)(pk.y & 0xffff)) * q0.z; e += bf2f((u16)(pk.y >> 16)) * q0.w;
        e += bf2f((u16)(pk.z & 0xffff)) * q1.x; e += bf2f((u16)(pk.z >> 16)) * q1.y;
        e += bf2f((u16)(pk.w & 0xffff)) * q1.z; e += bf2f((u16)(pk.w >> 16)) * q1.w;
      }
    }
    energy[b * TLEN + t] = e;
    mx = fmaxf(mx, e);
  }
  #pragma unroll
  for (int s = 16; s > 0; s >>= 1) mx = fmaxf(mx, __shfl_xor(mx, s, 32));

  float sum = 0.f;
  for (int t0 = 0; t0 < TLEN; t0 += 32) {
    int t = t0 + lane;
    float p = __expf(energy[b * TLEN + t] - mx);
    energy[b * TLEN + t] = p;
    sum += p;
  }
  #pragma unroll
  for (int s = 16; s > 0; s >>= 1) sum += __shfl_xor(sum, s, 32);
  float inv = 1.f / sum;

  // ctx: lanes cover v = lane*4 .. lane*4+3 (128 contiguous bf16 per row).
  float a0 = 0, a1 = 0, a2 = 0, a3 = 0;
  int v0 = lane * 4;
  #pragma unroll 4
  for (int t = 0; t < red; ++t) {       // att weights for t >= red are exactly 0
    float w = energy[b * TLEN + t];     // LDS broadcast (uniform address)
    const u16* vp = Vbf + ((size_t)t * NBATCH + n) * VSZN + v0;
    uint2 pv = *reinterpret_cast<const uint2*>(vp);
    a0 += w * bf2f((u16)(pv.x & 0xffff)); a1 += w * bf2f((u16)(pv.x >> 16));
    a2 += w * bf2f((u16)(pv.y & 0xffff)); a3 += w * bf2f((u16)(pv.y >> 16));
  }
  float* cp = ctx + b * VSZN + v0;
  cp[0] = a0 * inv; cp[1] = a1 * inv; cp[2] = a2 * inv; cp[3] = a3 * inv;
}

__global__ __launch_bounds__(NTHREADS)
void decoder_kernel(const u16* __restrict__ Kbf, const u16* __restrict__ Vbf,
                    const u16* __restrict__ W1S, const u16* __restrict__ W2S,
                    const u16* __restrict__ WqS, const u16* __restrict__ WfcS,
                    const u16* __restrict__ WeS,
                    const int* __restrict__ text, const int* __restrict__ redlen,
                    const float* __restrict__ emb,
                    const float* __restrict__ bq,
                    const float* __restrict__ bih1, const float* __restrict__ bhh1,
                    const float* __restrict__ bih2, const float* __restrict__ bhh2,
                    const float* __restrict__ bfc, const float* __restrict__ bout,
                    float* __restrict__ out) {
  extern __shared__ __align__(128) char smem[];
  float* energy = (float*)(smem);                 // 16*1024*4 = 65536
  float* h1     = (float*)(smem + 65536);         // 16*512*4  = 32768
  float* c1     = (float*)(smem + 98304);         // 32768
  float* h2     = (float*)(smem + 131072);        // 16*128*4  = 8192
  float* c2     = (float*)(smem + 139264);        // 8192
  float* cebuf  = (float*)(smem + 147456);        // 8192
  float* ctx1   = (float*)(smem + 155648);        // 8192
  float* ctx2   = (float*)(smem + 163840);        // 8192
  float* qbuf   = (float*)(smem + 172032);        // 8192
  float* outbuf = (float*)(smem + 180224);        // 8192
  u16*   Abuf   = (u16*)  (smem + 188416);        // 24*32*16*2 = 24576 -> 212992 total

  int tid = threadIdx.x, lane = tid & 31, wave = tid >> 5;  // 16 waves
  int nb = blockIdx.x * 16;

  for (int id = tid; id < 16 * HIDN; id += NTHREADS) { h1[id] = 0.f; c1[id] = 0.f; }
  for (int id = tid; id < 16 * KSZN; id += NTHREADS) { h2[id] = 0.f; c2[id] = 0.f; }
  __syncthreads();

  for (int i = 0; i < LSTEPS; ++i) {
    // P1: teacher-forced char embedding (step 0 -> padding row 0)
    for (int id = tid; id < 16 * EMBN; id += NTHREADS) {
      int m = id >> 7, e = id & 127;
      int cidx = (i == 0) ? 0 : text[(nb + m) * LSTEPS + (i - 1)];
      cebuf[id] = emb[cidx * EMBN + e];
    }
    __syncthreads();
    // P2/P3: q1 = ce @ Wq^T + bq
    swizzleA(Abuf, cebuf, EMBN, nullptr, 0, nullptr, 0, EMBN, tid);
    __syncthreads();
    if (wave < 8) {
      v8f acc = wmma_tile(Abuf, WqS, wave, 4, lane);
      int nn = wave * 16 + (lane & 15);
      float bb = bq[nn];
      #pragma unroll
      for (int r = 0; r < 8; ++r)
        qbuf[(r + ((lane >> 4) << 3)) * KSZN + nn] = acc[r] + bb;
    }
    __syncthreads();
    // P4: attention 1
    attend(Kbf, Vbf, redlen, qbuf, energy, ctx1, nb, wave, lane);
    __syncthreads();
    // P5/P6: LSTM1, x = [ce | ctx1], K = 256 + 512; 32 tiles over 16 waves
    swizzleA(Abuf, cebuf, EMBN, ctx1, VSZN, h1, HIDN, 768, tid);
    __syncthreads();
    for (int r2 = 0; r2 < 2; ++r2) {
      int jt = wave * 2 + r2;
      v8f g[4];
      #pragma unroll
      for (int G = 0; G < 4; ++G) g[G] = wmma_tile(Abuf, W1S, G * 32 + jt, 24, lane);
      int j = jt * 16 + (lane & 15);
      float bi  = bih1[j] + bhh1[j];
      float bf_ = bih1[512 + j] + bhh1[512 + j];
      float bg  = bih1[1024 + j] + bhh1[1024 + j];
      float bo  = bih1[1536 + j] + bhh1[1536 + j];
      #pragma unroll
      for (int r = 0; r < 8; ++r) {
        int m = r + ((lane >> 4) << 3);
        float cv = sigm(g[1][r] + bf_) * c1[m * HIDN + j]
                 + sigm(g[0][r] + bi) * tanh_fast(g[2][r] + bg);
        c1[m * HIDN + j] = cv;
        h1[m * HIDN + j] = sigm(g[3][r] + bo) * tanh_fast(cv);
      }
    }
    __syncthreads();
    // P7/P8: LSTM2, x = h1, K = 512 + 128
    swizzleA(Abuf, h1, HIDN, h2, KSZN, nullptr, 0, 640, tid);
    __syncthreads();
    if (wave < 8) {
      int jt = wave;
      v8f g[4];
      #pragma unroll
      for (int G = 0; G < 4; ++G) g[G] = wmma_tile(Abuf, W2S, G * 8 + jt, 20, lane);
      int j = jt * 16 + (lane & 15);
      float bi  = bih2[j] + bhh2[j];
      float bf_ = bih2[128 + j] + bhh2[128 + j];
      float bg  = bih2[256 + j] + bhh2[256 + j];
      float bo  = bih2[384 + j] + bhh2[384 + j];
      #pragma unroll
      for (int r = 0; r < 8; ++r) {
        int m = r + ((lane >> 4) << 3);
        float cv = sigm(g[1][r] + bf_) * c2[m * KSZN + j]
                 + sigm(g[0][r] + bi) * tanh_fast(g[2][r] + bg);
        c2[m * KSZN + j] = cv;
        h2[m * KSZN + j] = sigm(g[3][r] + bo) * tanh_fast(cv);
      }
    }
    __syncthreads();
    // P9/P10: q2 = h2 @ Wq^T + bq
    swizzleA(Abuf, h2, KSZN, nullptr, 0, nullptr, 0, KSZN, tid);
    __syncthreads();
    if (wave < 8) {
      v8f acc = wmma_tile(Abuf, WqS, wave, 4, lane);
      int nn = wave * 16 + (lane & 15);
      float bb = bq[nn];
      #pragma unroll
      for (int r = 0; r < 8; ++r)
        qbuf[(r + ((lane >> 4) << 3)) * KSZN + nn] = acc[r] + bb;
    }
    __syncthreads();
    // P11: attention 2
    attend(Kbf, Vbf, redlen, qbuf, energy, ctx2, nb, wave, lane);
    __syncthreads();
    // P12/P13: fc + hardtanh
    swizzleA(Abuf, h2, KSZN, ctx2, VSZN, nullptr, 0, 256, tid);
    __syncthreads();
    if (wave < 8) {
      v8f acc = wmma_tile(Abuf, WfcS, wave, 8, lane);
      int nn = wave * 16 + (lane & 15);
      float bb = bfc[nn];
      #pragma unroll
      for (int r = 0; r < 8; ++r) {
        float o = acc[r] + bb;
        o = fminf(1.f, fmaxf(-1.f, o));
        outbuf[(r + ((lane >> 4) << 3)) * EMBN + nn] = o;
      }
    }
    __syncthreads();
    // P14/P15: pred = out @ emb^T + b_out  -> global
    swizzleA(Abuf, outbuf, EMBN, nullptr, 0, nullptr, 0, EMBN, tid);
    __syncthreads();
    if (wave < 3) {
      v8f acc = wmma_tile(Abuf, WeS, wave, 4, lane);
      int v = wave * 16 + (lane & 15);
      if (v < VOCABN) {
        float bb = bout[v];
        #pragma unroll
        for (int r = 0; r < 8; ++r) {
          int m = r + ((lane >> 4) << 3);
          out[((size_t)(nb + m) * LSTEPS + i) * VOCABN + v] = acc[r] + bb;
        }
      }
    }
    __syncthreads();
  }
}

// ---- one-time prep: fp32 -> bf16 key/values; weights -> bf16 B-operand swizzle ----
__global__ void prep_convert(const float* __restrict__ kp, const float* __restrict__ vv,
                             u16* __restrict__ Kbf, u16* __restrict__ Vbf, int n) {
  for (int id = blockIdx.x * blockDim.x + threadIdx.x; id < 2 * n;
       id += gridDim.x * blockDim.x) {
    if (id < n) Kbf[id] = f2bf(kp[id]);
    else        Vbf[id - n] = f2bf(vv[id - n]);
  }
}

// B element (k, n): n = lane&15, k = 32*kt + 16*(j/8) + 8*(lane/16) + (j%8)
#define SWZ_DECODE(KT) \
  int j = id & 15, lane = (id >> 4) & 31, kt = (id >> 9) % (KT), nt = id / ((KT) * 512); \
  int n = nt * 16 + (lane & 15); \
  int k = kt * 32 + ((j >> 3) << 4) + ((lane >> 4) << 3) + (j & 7);

__global__ void prep_weights(const float* __restrict__ Wq,
                             const float* __restrict__ Wih1, const float* __restrict__ Whh1,
                             const float* __restrict__ Wih2, const float* __restrict__ Whh2,
                             const float* __restrict__ Wfc,  const float* __restrict__ emb,
                             u16* __restrict__ WqS, u16* __restrict__ W1S,
                             u16* __restrict__ W2S, u16* __restrict__ WfcS,
                             u16* __restrict__ WeS) {
  int tid0 = blockIdx.x * blockDim.x + threadIdx.x;
  int stride = gridDim.x * blockDim.x;
  for (int id = tid0; id < 128 * 24 * 512; id += stride) {  // [Wih1|Whh1], K=768, O=2048
    SWZ_DECODE(24)
    float v = (k < 256) ? Wih1[n * 256 + k] : Whh1[n * 512 + (k - 256)];
    W1S[id] = f2bf(v);
  }
  for (int id = tid0; id < 32 * 20 * 512; id += stride) {   // [Wih2|Whh2], K=640, O=512
    SWZ_DECODE(20)
    float v = (k < 512) ? Wih2[n * 512 + k] : Whh2[n * 128 + (k - 512)];
    W2S[id] = f2bf(v);
  }
  for (int id = tid0; id < 8 * 4 * 512; id += stride) {     // Wq, K=128, O=128
    SWZ_DECODE(4)
    WqS[id] = f2bf(Wq[n * 128 + k]);
  }
  for (int id = tid0; id < 8 * 8 * 512; id += stride) {     // Wfc, K=256, O=128
    SWZ_DECODE(8)
    WfcS[id] = f2bf(Wfc[n * 256 + k]);
  }
  for (int id = tid0; id < 3 * 4 * 512; id += stride) {     // emb^T, K=128, O=34 (pad 48)
    SWZ_DECODE(4)
    WeS[id] = f2bf((n < VOCABN) ? emb[n * 128 + k] : 0.f);
  }
}

extern "C" void kernel_launch(void* const* d_in, const int* in_sizes, int n_in,
                              void* d_out, int out_size, void* d_ws, size_t ws_size,
                              hipStream_t stream) {
  (void)in_sizes; (void)n_in; (void)out_size; (void)ws_size;
  const float* key  = (const float*)d_in[0];
  const float* val  = (const float*)d_in[1];
  const int*   red  = (const int*)d_in[2];
  const int*   text = (const int*)d_in[3];
  const float* emb  = (const float*)d_in[4];
  const float* Wq   = (const float*)d_in[5];
  const float* bq   = (const float*)d_in[6];
  const float* Wih1 = (const float*)d_in[7];
  const float* Whh1 = (const float*)d_in[8];
  const float* bih1 = (const float*)d_in[9];
  const float* bhh1 = (const float*)d_in[10];
  const float* Wih2 = (const float*)d_in[11];
  const float* Whh2 = (const float*)d_in[12];
  const float* bih2 = (const float*)d_in[13];
  const float* bhh2 = (const float*)d_in[14];
  const float* Wfc  = (const float*)d_in[15];
  const float* bfc  = (const float*)d_in[16];
  const float* bout = (const float*)d_in[17];

  u16* ws = (u16*)d_ws;
  const size_t KVN = (size_t)TLEN * NBATCH * KSZN;  // 16,777,216
  u16* Kbf  = ws;
  u16* Vbf  = Kbf + KVN;
  u16* W1S  = Vbf + KVN;
  u16* W2S  = W1S + (size_t)128 * 24 * 512;
  u16* WqS  = W2S + (size_t)32 * 20 * 512;
  u16* WfcS = WqS + (size_t)8 * 4 * 512;
  u16* WeS  = WfcS + (size_t)8 * 8 * 512;

  hipLaunchKernelGGL(prep_convert, dim3(2048), dim3(256), 0, stream,
                     key, val, Kbf, Vbf, (int)KVN);
  hipLaunchKernelGGL(prep_weights, dim3(1024), dim3(256), 0, stream,
                     Wq, Wih1, Whh1, Wih2, Whh2, Wfc, emb,
                     WqS, W1S, W2S, WfcS, WeS);

  const int SHMEM = 212992;
  (void)hipFuncSetAttribute(reinterpret_cast<const void*>(decoder_kernel),
                            hipFuncAttributeMaxDynamicSharedMemorySize, SHMEM);
  hipLaunchKernelGGL(decoder_kernel, dim3(NBATCH / 16), dim3(NTHREADS), SHMEM, stream,
                     Kbf, Vbf, W1S, W2S, WqS, WfcS, WeS,
                     text, red, emb, bq, bih1, bhh1, bih2, bhh2, bfc, bout,
                     (float*)d_out);
}
